// TransformerBlock_2001454759955
// MI455X (gfx1250) — matrix operand order repaired and checked by reading, not compiled, not run on hardware
//
#include <hip/hip_runtime.h>
#include <cstdint>

typedef __attribute__((ext_vector_type(16))) _Float16 v16h;
typedef __attribute__((ext_vector_type(8)))  _Float16 v8h;
typedef __attribute__((ext_vector_type(8)))  float    v8f;

static __device__ __forceinline__ v8f wmma_f32_f16(v16h a, v16h b, v8f c) {
  // D = A(16x32,f16) * B(32x16,f16) + C(16x16,f32)
  return __builtin_amdgcn_wmma_f32_16x16x32_f16(false, a, false, b, (short)0, c,
                                                false, false);
}

// CDNA5 async global->LDS copy (ASYNCcnt-tracked, no VGPR round trip).
// lds: byte offset of destination within workgroup LDS (low 32 bits of the
// flat shared-pointer are the LDS offset per the aperture mapping).
static __device__ __forceinline__ void async_load_b128(uint32_t lds,
                                                       const void* gptr) {
  asm volatile("global_load_async_to_lds_b128 %0, %1, off"
               :
               : "v"(lds), "v"(gptr)
               : "memory");
}
static __device__ __forceinline__ void wait_async0() {
  asm volatile("s_wait_asynccnt 0x0" ::: "memory");
}
static __device__ __forceinline__ uint32_t lds_addr(const void* p) {
  return (uint32_t)(uintptr_t)p;
}

// ---------------------------------------------------------------------------
// Elementwise helpers
// ---------------------------------------------------------------------------
__global__ void cvt_f32_f16(const float* __restrict__ s, _Float16* __restrict__ d,
                            size_t n) {
  size_t i = (size_t)blockIdx.x * blockDim.x + threadIdx.x;
  if (i < n) d[i] = (_Float16)s[i];
}

__global__ void add_store(const float* __restrict__ x, const float* __restrict__ a,
                          float* __restrict__ second, float* __restrict__ out,
                          size_t n) {
  size_t i = (size_t)blockIdx.x * blockDim.x + threadIdx.x;
  if (i < n) {
    float v = x[i] + a[i];
    second[i] = v;
    out[i] = v;  // final = second + moe; moe accumulated later
  }
}

__global__ void silu_mul(const float* __restrict__ t1, const float* __restrict__ t3,
                         _Float16* __restrict__ o, size_t n) {
  size_t i = (size_t)blockIdx.x * blockDim.x + threadIdx.x;
  if (i < n) {
    float g = t1[i];
    float s = g / (1.0f + __expf(-g));
    o[i] = (_Float16)(s * t3[i]);
  }
}

__global__ void moe_accum(const float* __restrict__ y, const float* __restrict__ wfull,
                          float* __restrict__ out, int D, int e, size_t n) {
  size_t i = (size_t)blockIdx.x * blockDim.x + threadIdx.x;
  if (i < n) {
    size_t t = i / (size_t)D;
    out[i] += wfull[t * 8 + e] * y[i];
  }
}

// ---------------------------------------------------------------------------
// RMSNorm: fp32 in -> f16 out (row per block)
// ---------------------------------------------------------------------------
__global__ __launch_bounds__(256) void rmsnorm_f16(const float* __restrict__ x,
                                                   const float* __restrict__ w,
                                                   _Float16* __restrict__ out,
                                                   int C) {
  __shared__ float red[256];
  const int row = blockIdx.x;
  const float* xp = x + (size_t)row * C;
  float ss = 0.0f;
  for (int c = threadIdx.x; c < C; c += 256) {
    float v = xp[c];
    ss += v * v;
  }
  red[threadIdx.x] = ss;
  __syncthreads();
  for (int s = 128; s > 0; s >>= 1) {
    if ((int)threadIdx.x < s) red[threadIdx.x] += red[threadIdx.x + s];
    __syncthreads();
  }
  const float inv = rsqrtf(red[0] / (float)C + 1e-6f);
  for (int c = threadIdx.x; c < C; c += 256)
    out[(size_t)row * C + c] = (_Float16)(xp[c] * inv * w[c]);
}

// ---------------------------------------------------------------------------
// Tiled WMMA GEMM: C[M,N] = A[M,K] (f16, row-major) * W[N,K]^T (f16, row-major)
// Block tile 128x128, K-tile 32, double-buffered LDS staged with
// global_load_async_to_lds_b128. 8 waves: 4x2, each wave 32x64 (2x4 WMMA).
// Requires M%128==0, N%128==0, K%32==0 (true for all shapes here).
// ---------------------------------------------------------------------------
#define BM 128
#define BN 128
#define BK 32

__global__ __launch_bounds__(256) void gemm_f16_wmma(
    const _Float16* __restrict__ A, const _Float16* __restrict__ W,
    float* __restrict__ Cf, _Float16* __restrict__ Ch, int M, int N, int K,
    int storeHalf) {
  __shared__ __align__(16) _Float16 As[2][BM * BK];
  __shared__ __align__(16) _Float16 Bs[2][BN * BK];
  const int m0 = blockIdx.y * BM;
  const int n0 = blockIdx.x * BN;
  const int tid = threadIdx.x;
  const int lane = tid & 31;
  const int wave = tid >> 5;
  const int wrow = wave >> 1;  // 0..3 -> 32 rows each
  const int wcol = wave & 1;   // 0..1 -> 64 cols each
  const int g = lane >> 4;
  const int r = lane & 15;

  v8f acc[2][4];
#pragma unroll
  for (int mi = 0; mi < 2; ++mi)
#pragma unroll
    for (int ni = 0; ni < 4; ++ni)
#pragma unroll
      for (int v = 0; v < 8; ++v) acc[mi][ni][v] = 0.0f;

  const int arow = tid >> 1;        // 0..127
  const int acol = (tid & 1) * 16;  // 0 or 16
  const _Float16* gA = A + (size_t)(m0 + arow) * K + acol;
  const _Float16* gW = W + (size_t)(n0 + arow) * K + acol;

  auto stage = [&](int buf, int k0) {
    const uint32_t la = lds_addr(&As[buf][arow * BK + acol]);
    const uint32_t lb = lds_addr(&Bs[buf][arow * BK + acol]);
    async_load_b128(la, gA + k0);
    async_load_b128(la + 16, gA + k0 + 8);
    async_load_b128(lb, gW + k0);
    async_load_b128(lb + 16, gW + k0 + 8);
  };

  stage(0, 0);
  wait_async0();
  __syncthreads();

  int cur = 0;
  for (int k0 = 0; k0 < K; k0 += BK) {
    if (k0 + BK < K) stage(cur ^ 1, k0 + BK);  // overlap with WMMA below

    v16h af[2], bf[4];
#pragma unroll
    for (int mi = 0; mi < 2; ++mi) {
      const _Float16* ap = &As[cur][(wrow * 32 + mi * 16 + r) * BK];
#pragma unroll
      for (int j = 0; j < 8; ++j) {
        af[mi][j] = ap[16 * g + j];           // K = 16g..16g+7
        af[mi][8 + j] = ap[16 * g + 8 + j];  // K = 16g+8..16g+15
      }
    }
#pragma unroll
    for (int ni = 0; ni < 4; ++ni) {
      const _Float16* bp = &Bs[cur][(wcol * 64 + ni * 16 + r) * BK];  // col n = r
#pragma unroll
      for (int j = 0; j < 16; ++j) bf[ni][j] = bp[16 * g + j];  // K = 16g+j
    }
#pragma unroll
    for (int mi = 0; mi < 2; ++mi)
#pragma unroll
      for (int ni = 0; ni < 4; ++ni)
        acc[mi][ni] = wmma_f32_f16(af[mi], bf[ni], acc[mi][ni]);

    wait_async0();    // next tile landed in LDS (this wave's transfers)
    __syncthreads();  // ...and everyone else's; reads of cur done block-wide
    cur ^= 1;
  }

  // C layout: lane holds (M = 8g+v, N = r) for v = 0..7
#pragma unroll
  for (int mi = 0; mi < 2; ++mi) {
#pragma unroll
    for (int ni = 0; ni < 4; ++ni) {
      const int mbase = m0 + wrow * 32 + mi * 16 + 8 * g;
      const int nc = n0 + wcol * 64 + ni * 16 + r;
#pragma unroll
      for (int v = 0; v < 8; ++v) {
        const size_t idx = (size_t)(mbase + v) * N + nc;
        if (storeHalf)
          Ch[idx] = (_Float16)acc[mi][ni][v];
        else
          Cf[idx] = acc[mi][ni][v];
      }
    }
  }
}

// ---------------------------------------------------------------------------
// Flash attention (causal), WMMA for QK^T and PV, async-to-LDS staging.
// grid: (S/64, B*H), block: 128 (4 waves, 16 q-rows each). Hd = 128.
// Q: [B*S, H*128] f16 ; KV: [B*S, H*256] f16 (per head: 128 k, 128 v)
// ---------------------------------------------------------------------------
__global__ __launch_bounds__(128) void mla_flash_attn(
    const _Float16* __restrict__ Q, const _Float16* __restrict__ KV,
    _Float16* __restrict__ O, int B, int S, int H) {
  const int HD = 128;
  __shared__ __align__(16) _Float16 Qs[64 * 128];
  __shared__ __align__(16) _Float16 Ks[32 * 128];
  __shared__ __align__(16) _Float16 Vs[32 * 128];
  __shared__ __align__(16) _Float16 Ps[4][16 * 32];

  const int bh = blockIdx.y;
  const int b = bh / H, h = bh % H;
  const int q0 = blockIdx.x * 64;
  const int tid = threadIdx.x;
  const int lane = tid & 31, wave = tid >> 5;
  const int g = lane >> 4, r = lane & 15;

  const size_t qstride = (size_t)H * HD;       // 2048
  const size_t kvstride = (size_t)H * 2 * HD;  // 4096
  const _Float16* Qb = Q + (size_t)b * S * qstride + (size_t)h * HD;
  const _Float16* Kb = KV + (size_t)b * S * kvstride + (size_t)h * 2 * HD;
  const _Float16* Vb = Kb + HD;

  // Q tile 64x128, staged asynchronously once (wait folded into first iter)
  for (int i = tid; i < 64 * 16; i += 128) {
    const int row = i >> 4, c8 = i & 15;
    async_load_b128(lds_addr(Qs + row * 128 + c8 * 8),
                    Qb + (size_t)(q0 + row) * qstride + c8 * 8);
  }

  v8f o_acc[8];
#pragma unroll
  for (int nt = 0; nt < 8; ++nt)
#pragma unroll
    for (int v = 0; v < 8; ++v) o_acc[nt][v] = 0.0f;
  float m_i[8], l_i[8];
#pragma unroll
  for (int v = 0; v < 8; ++v) {
    m_i[v] = -1e30f;
    l_i[v] = 0.0f;
  }

  const float scale = 0.088388347648318447f;  // 1/sqrt(128)
  const int kend = q0 + 64;

  for (int key0 = 0; key0 < kend; key0 += 32) {
    __syncthreads();  // all reads of Ks/Vs from previous iteration done
    for (int i = tid; i < 32 * 16; i += 128) {
      const int row = i >> 4, c8 = i & 15;
      async_load_b128(lds_addr(Ks + row * 128 + c8 * 8),
                      Kb + (size_t)(key0 + row) * kvstride + c8 * 8);
      async_load_b128(lds_addr(Vs + row * 128 + c8 * 8),
                      Vb + (size_t)(key0 + row) * kvstride + c8 * 8);
    }
    wait_async0();
    __syncthreads();

    // scores: Q(16x128) x K^T -> two 16x16 tiles (keys r and 16+r)
    v8f sc0, sc1;
#pragma unroll
    for (int v = 0; v < 8; ++v) {
      sc0[v] = 0.0f;
      sc1[v] = 0.0f;
    }
    for (int kk = 0; kk < 128; kk += 32) {
      v16h aq;
      const _Float16* ap = Qs + (wave * 16 + r) * 128 + kk;
#pragma unroll
      for (int j = 0; j < 8; ++j) {
        aq[j] = ap[16 * g + j];
        aq[8 + j] = ap[16 * g + 8 + j];
      }
      v16h bk0, bk1;
      const _Float16* bp0 = Ks + r * 128 + kk;         // key r
      const _Float16* bp1 = Ks + (16 + r) * 128 + kk;  // key 16+r
#pragma unroll
      for (int j = 0; j < 16; ++j) {
        bk0[j] = bp0[16 * g + j];
        bk1[j] = bp1[16 * g + j];
      }
      sc0 = wmma_f32_f16(aq, bk0, sc0);
      sc1 = wmma_f32_f16(aq, bk1, sc1);
    }

    // online softmax: lane owns rows m=8g+v, cols r (tile0) and 16+r (tile1)
    float corr[8];
#pragma unroll
    for (int v = 0; v < 8; ++v) {
      const int qg = q0 + wave * 16 + 8 * g + v;
      float s0 = sc0[v] * scale;
      if (key0 + r > qg) s0 = -1e30f;
      float s1 = sc1[v] * scale;
      if (key0 + 16 + r > qg) s1 = -1e30f;
      float mx = fmaxf(s0, s1);
#pragma unroll
      for (int off = 1; off < 16; off <<= 1) mx = fmaxf(mx, __shfl_xor(mx, off, 32));
      const float mnew = fmaxf(m_i[v], mx);
      const float e0 = __expf(s0 - mnew);
      const float e1 = __expf(s1 - mnew);
      Ps[wave][(8 * g + v) * 32 + r] = (_Float16)e0;
      Ps[wave][(8 * g + v) * 32 + 16 + r] = (_Float16)e1;
      float ps = e0 + e1;
#pragma unroll
      for (int off = 1; off < 16; off <<= 1) ps += __shfl_xor(ps, off, 32);
      corr[v] = __expf(m_i[v] - mnew);
      l_i[v] = l_i[v] * corr[v] + ps;
      m_i[v] = mnew;
    }
#pragma unroll
    for (int nt = 0; nt < 8; ++nt)
#pragma unroll
      for (int v = 0; v < 8; ++v) o_acc[nt][v] *= corr[v];
    __syncthreads();  // Ps visible across lanes before fragment reload

    // O += P(16x32) x V(32x128)
    v16h apf;
    const _Float16* pp = Ps[wave] + r * 32;
#pragma unroll
    for (int j = 0; j < 8; ++j) {
      apf[j] = pp[16 * g + j];
      apf[8 + j] = pp[16 * g + 8 + j];
    }
#pragma unroll
    for (int nt = 0; nt < 8; ++nt) {
      v16h bv;
#pragma unroll
      for (int j = 0; j < 16; ++j) bv[j] = Vs[(16 * g + j) * 128 + nt * 16 + r];
      o_acc[nt] = wmma_f32_f16(apf, bv, o_acc[nt]);
    }
  }

#pragma unroll
  for (int nt = 0; nt < 8; ++nt) {
#pragma unroll
    for (int v = 0; v < 8; ++v) {
      const int row = q0 + wave * 16 + 8 * g + v;
      const size_t tok = (size_t)b * S + row;
      O[tok * qstride + (size_t)h * HD + nt * 16 + r] =
          (_Float16)(o_acc[nt][v] / l_i[v]);
    }
  }
}

// ---------------------------------------------------------------------------
// MoE gating: softmax(h @ gate_w^T) -> top-2 renormalized, dense per-expert wts
// One wave per token.
// ---------------------------------------------------------------------------
__global__ __launch_bounds__(32) void gate_topk(const _Float16* __restrict__ h2,
                                                const float* __restrict__ gw,
                                                float* __restrict__ wfull, int D) {
  const int t = blockIdx.x;
  const int lane = threadIdx.x;
  float acc[8];
#pragma unroll
  for (int e = 0; e < 8; ++e) acc[e] = 0.0f;
  const _Float16* hp = h2 + (size_t)t * D;
  for (int d = lane; d < D; d += 32) {
    const float hv = (float)hp[d];
#pragma unroll
    for (int e = 0; e < 8; ++e) acc[e] += hv * gw[(size_t)e * D + d];
  }
#pragma unroll
  for (int e = 0; e < 8; ++e)
#pragma unroll
    for (int off = 16; off > 0; off >>= 1) acc[e] += __shfl_xor(acc[e], off, 32);
  if (lane == 0) {
    int i1 = 0;
    for (int e = 1; e < 8; ++e)
      if (acc[e] > acc[i1]) i1 = e;
    int i2 = -1;
    for (int e = 0; e < 8; ++e) {
      if (e == i1) continue;
      if (i2 < 0 || acc[e] > acc[i2]) i2 = e;
    }
    const float a = acc[i1], bb = acc[i2];
    const float m = fmaxf(a, bb);
    const float ea = __expf(a - m), eb = __expf(bb - m);
    const float s = ea + eb;
    for (int e = 0; e < 8; ++e) wfull[(size_t)t * 8 + e] = 0.0f;
    wfull[(size_t)t * 8 + i1] = ea / s;
    wfull[(size_t)t * 8 + i2] = eb / s;
  }
}

// ---------------------------------------------------------------------------
// Launch
// ---------------------------------------------------------------------------
extern "C" void kernel_launch(void* const* d_in, const int* in_sizes, int n_in,
                              void* d_out, int out_size, void* d_ws, size_t ws_size,
                              hipStream_t stream) {
  (void)in_sizes; (void)n_in; (void)out_size; (void)ws_size;
  constexpr int B = 2, S = 2048, D = 2048, H = 16, Hd = 128;
  constexpr int QR = 1024, KVR = 512, E = 8, F = 2048;
  constexpr int NT = B * S;  // 4096 tokens

  const float* x = (const float*)d_in[0];
  const float* norm_w = (const float*)d_in[1];
  const float* wq_a = (const float*)d_in[2];
  const float* q_norm = (const float*)d_in[3];
  const float* wq_b = (const float*)d_in[4];
  const float* wkv_a = (const float*)d_in[5];
  const float* kv_norm = (const float*)d_in[6];
  const float* wkv_b = (const float*)d_in[7];
  const float* wo = (const float*)d_in[8];
  const float* gate_w = (const float*)d_in[9];
  const float* w1 = (const float*)d_in[10];
  const float* w3 = (const float*)d_in[11];
  const float* w2 = (const float*)d_in[12];
  float* out = (float*)d_out;

  char* ws = (char*)d_ws;
  size_t off = 0;
  auto alloc = [&](size_t bytes) -> void* {
    void* p = ws + off;
    off = (off + bytes + 255) & ~(size_t)255;
    return p;
  };

  _Float16* wqa_h = (_Float16*)alloc((size_t)QR * D * 2);
  _Float16* wqb_h = (_Float16*)alloc((size_t)H * Hd * QR * 2);
  _Float16* wkva_h = (_Float16*)alloc((size_t)KVR * D * 2);
  _Float16* wkvb_h = (_Float16*)alloc((size_t)H * 2 * Hd * KVR * 2);
  _Float16* wo_h = (_Float16*)alloc((size_t)D * H * Hd * 2);
  _Float16* w1_h = (_Float16*)alloc((size_t)E * F * D * 2);
  _Float16* w3_h = (_Float16*)alloc((size_t)E * F * D * 2);
  _Float16* w2_h = (_Float16*)alloc((size_t)E * D * F * 2);
  _Float16* h1_h = (_Float16*)alloc((size_t)NT * D * 2);
  float* qlat_f = (float*)alloc((size_t)NT * QR * 4);
  _Float16* qlat_h = (_Float16*)alloc((size_t)NT * QR * 2);
  _Float16* q_h = (_Float16*)alloc((size_t)NT * H * Hd * 2);
  float* kvlat_f = (float*)alloc((size_t)NT * KVR * 4);
  _Float16* kvlat_h = (_Float16*)alloc((size_t)NT * KVR * 2);
  _Float16* kv_h = (_Float16*)alloc((size_t)NT * H * 2 * Hd * 2);
  _Float16* ao_h = (_Float16*)alloc((size_t)NT * D * 2);
  float* attn_f = (float*)alloc((size_t)NT * D * 4);
  float* second = (float*)alloc((size_t)NT * D * 4);
  _Float16* h2_h = (_Float16*)alloc((size_t)NT * D * 2);
  float* wfull = (float*)alloc((size_t)NT * E * 4);
  float* t1 = (float*)alloc((size_t)NT * F * 4);
  float* t3 = (float*)alloc((size_t)NT * F * 4);
  _Float16* gu_h = (_Float16*)alloc((size_t)NT * F * 2);
  float* yb = (float*)alloc((size_t)NT * D * 4);

  auto cvt = [&](const float* s, _Float16* d, size_t n) {
    cvt_f32_f16<<<dim3((unsigned)((n + 255) / 256)), dim3(256), 0, stream>>>(s, d, n);
  };
  auto gemm = [&](const _Float16* A, const _Float16* W, float* Cf, _Float16* Ch,
                  int M, int N, int K, int sh) {
    gemm_f16_wmma<<<dim3(N / 128, M / 128), dim3(256), 0, stream>>>(A, W, Cf, Ch,
                                                                    M, N, K, sh);
  };

  // weight precision conversion (one HBM pass, ~30us at 23.3 TB/s)
  cvt(wq_a, wqa_h, (size_t)QR * D);
  cvt(wq_b, wqb_h, (size_t)H * Hd * QR);
  cvt(wkv_a, wkva_h, (size_t)KVR * D);
  cvt(wkv_b, wkvb_h, (size_t)H * 2 * Hd * KVR);
  cvt(wo, wo_h, (size_t)D * H * Hd);
  cvt(w1, w1_h, (size_t)E * F * D);
  cvt(w3, w3_h, (size_t)E * F * D);
  cvt(w2, w2_h, (size_t)E * D * F);

  // MLA attention
  rmsnorm_f16<<<NT, 256, 0, stream>>>(x, norm_w, h1_h, D);
  gemm(h1_h, wqa_h, qlat_f, nullptr, NT, QR, D, 0);
  rmsnorm_f16<<<NT, 256, 0, stream>>>(qlat_f, q_norm, qlat_h, QR);
  gemm(qlat_h, wqb_h, nullptr, q_h, NT, H * Hd, QR, 1);
  gemm(h1_h, wkva_h, kvlat_f, nullptr, NT, KVR, D, 0);
  rmsnorm_f16<<<NT, 256, 0, stream>>>(kvlat_f, kv_norm, kvlat_h, KVR);
  gemm(kvlat_h, wkvb_h, nullptr, kv_h, NT, H * 2 * Hd, KVR, 1);
  mla_flash_attn<<<dim3(S / 64, B * H), dim3(128), 0, stream>>>(q_h, kv_h, ao_h,
                                                                B, S, H);
  gemm(ao_h, wo_h, attn_f, nullptr, NT, D, H * Hd, 0);
  add_store<<<((size_t)NT * D + 255) / 256, 256, 0, stream>>>(x, attn_f, second,
                                                              out, (size_t)NT * D);

  // MoE
  rmsnorm_f16<<<NT, 256, 0, stream>>>(second, norm_w, h2_h, D);
  gate_topk<<<NT, 32, 0, stream>>>(h2_h, gate_w, wfull, D);
  for (int e = 0; e < E; ++e) {
    gemm(h2_h, w1_h + (size_t)e * F * D, t1, nullptr, NT, F, D, 0);
    gemm(h2_h, w3_h + (size_t)e * F * D, t3, nullptr, NT, F, D, 0);
    silu_mul<<<((size_t)NT * F + 255) / 256, 256, 0, stream>>>(t1, t3, gu_h,
                                                               (size_t)NT * F);
    gemm(gu_h, w2_h + (size_t)e * D * F, yb, nullptr, NT, D, F, 0);
    moe_accum<<<((size_t)NT * D + 255) / 256, 256, 0, stream>>>(
        yb, wfull, out, D, e, (size_t)NT * D);
  }
}
